// CapsNet_25194278158439
// MI455X (gfx1250) — matrix operand order repaired and checked by reading, not compiled, not run on hardware
//
#include <hip/hip_runtime.h>
#include <hip/hip_bf16.h>
#include <hip/hip_fp16.h>

typedef __attribute__((ext_vector_type(16))) _Float16     v16h_t;
typedef __attribute__((ext_vector_type(8)))  _Float16     v8h_t;
typedef __attribute__((ext_vector_type(8)))  float        v8f_t;
typedef __attribute__((ext_vector_type(4)))  unsigned int v4u_t;
typedef __attribute__((ext_vector_type(8)))  int          v8i_t;
typedef __attribute__((ext_vector_type(4)))  int          v4i_t;

#define NUM_CLASSES 276
#define HW          576     // 24*24
#define P_ROUTES    18432   // 32*576
#define CE          4416    // 276*16
#define K2          20736   // 81*256
#define M2          1152    // 2*576

// ---------------- pack kernels ----------------

// conv1 weights [256,1,9,9] -> w1t[i=ky*9+kx][oc]  (coalesced in oc)
__global__ void pack_w1t(const float* __restrict__ w1, float* __restrict__ w1t) {
    int idx = blockIdx.x * 256 + threadIdx.x;           // 81*256
    int i = idx >> 8, oc = idx & 255;
    w1t[i * 256 + oc] = w1[oc * 81 + i];
}

// conv2 weights [256,256,9,9] -> Bpack f16 [oc][k], k=(ky*9+kx)*256+ic
__global__ void pack_w2(const float* __restrict__ w2, _Float16* __restrict__ Bp) {
    long idx = (long)blockIdx.x * 256 + threadIdx.x;    // 256*20736
    int oc = (int)(idx / K2), k = (int)(idx % K2);
    int khk = k >> 8, ic = k & 255;
    Bp[idx] = (_Float16)w2[((long)oc * 256 + ic) * 81 + khk];
}

// W [32,276,8,16] -> Wf f32 [t][d][cc*16+e]
__global__ void pack_Wf(const float* __restrict__ W, float* __restrict__ Wf) {
    long idx = (long)blockIdx.x * 256 + threadIdx.x;    // 32*8*4416
    int t = (int)(idx / (8 * CE));
    int r = (int)(idx % (8 * CE));
    int d = r / CE, ce = r % CE;
    int cc = ce >> 4, e = ce & 15;
    Wf[idx] = W[(((long)t * NUM_CLASSES + cc) * 8 + d) * 16 + e];
}

// ---------------- conv1 + relu, channels-last f16 ----------------
__global__ __launch_bounds__(256) void conv1_relu(const float* __restrict__ x,
                                                  const float* __restrict__ w1t,
                                                  const float* __restrict__ b1,
                                                  _Float16* __restrict__ H1) {
    __shared__ float patch[81];
    int blk = blockIdx.x;                  // 2*3136
    int b = blk / 3136, s = blk % 3136;
    int oy = s / 56, ox = s % 56;
    if (threadIdx.x < 81) {
        int ky = threadIdx.x / 9, kx = threadIdx.x % 9;
        patch[threadIdx.x] = x[((long)b * 64 + oy + ky) * 64 + ox + kx];
    }
    __syncthreads();
    int oc = threadIdx.x;
    float acc = b1[oc];
#pragma unroll 9
    for (int i = 0; i < 81; i++) acc = fmaf(patch[i], w1t[i * 256 + oc], acc);
    acc = fmaxf(acc, 0.f);
    H1[((long)(b * 56 + oy) * 56 + ox) * 256 + oc] = (_Float16)acc;
}

// ---------------- TDM: stage one 16-row x 256-elem f16 tap tile into LDS ----------
// 2D descriptor per CDNA5 ISA (count=1, type=2, data_size=2B,
// tile_dim0=256, tile_dim1=16, tensor_dim0_stride=K2)
__device__ __forceinline__ void tdm_load_tile(const _Float16* gsrc, unsigned ldsOff) {
    unsigned long long ga = (unsigned long long)(uintptr_t)gsrc;
    v4u_t g0;
    g0[0] = 1u;                                                // count=1, user mode
    g0[1] = ldsOff;                                            // lds_addr (bytes)
    g0[2] = (unsigned)ga;                                      // global_addr[31:0]
    g0[3] = (unsigned)((ga >> 32) & 0x1FFFFFFu) | (2u << 30);  // addr[56:32] | type=2
    v8i_t g1;
    g1[0] = (int)(1u << 16);        // workgroup_mask=0, data_size=1 (2 bytes)
    g1[1] = (int)(256u << 16);      // tensor_dim0[15:0]=256 in bits[31:16]
    g1[2] = (int)(16u << 16);       // tensor_dim0[31:16]=0, tensor_dim1[15:0]=16
    g1[3] = (int)(256u << 16);      // tensor_dim1[31:16]=0, tile_dim0=256
    g1[4] = 16;                     // tile_dim1=16, tile_dim2=0
    g1[5] = (int)20736u;            // tensor_dim0_stride[31:0] = K2
    g1[6] = 0;                      // stride[47:32]=0, tensor_dim1_stride lo=0
    g1[7] = 0;
    v4i_t z4 = {0, 0, 0, 0};
    v8i_t z8 = {0, 0, 0, 0, 0, 0, 0, 0};
    __builtin_amdgcn_tensor_load_to_lds(g0, g1, z4, z4, z8, 0);  // 6-arg (clang-23 form)
}

// ---------------- conv2: implicit GEMM, v_wmma_f32_16x16x32_f16 ----------------
// Grid: 16 nt x 18 mt-groups x 3 K-splits; block = 4 waves sharing one B tap tile
// (TDM -> LDS, double buffered), each wave owns one 16x16 C tile of a K-partial.
__global__ __launch_bounds__(128) void conv2_wmma(const _Float16* __restrict__ H1,
                                                  const _Float16* __restrict__ Bp,
                                                  float* __restrict__ O2p) {
    __shared__ __align__(32) _Float16 smemB[2][16 * 256];     // 2 x 8 KB
    int wid = threadIdx.x >> 5, lane = threadIdx.x & 31;
    int blk = blockIdx.x;
    int nt  = blk & 15;
    int r   = blk >> 4;                  // 0..53
    int mtg = r % 18, ks = r / 18;       // ks: K split 0..2 (taps ks*27 .. +27)
    int mt  = mtg * 4 + wid;             // 0..71
    int g = lane >> 4, ln = lane & 15;

    int m = mt * 16 + ln;                // this lane's A row
    int b = m / HW, s2 = m % HW;
    int oy = s2 / 24, ox = s2 % 24;
    const _Float16* aBase = H1 + ((long)(b * 56 + 2 * oy) * 56 + 2 * ox) * 256 + g * 8;

    const _Float16* Btap0 = Bp + (long)(nt * 16) * K2;   // 16 oc rows start here
    int tapBase = ks * 27;                               // 27 taps = 3 full ky rows
    unsigned ldsBase = (unsigned)(uintptr_t)&smemB[0][0];

    if (wid == 0) tdm_load_tile(Btap0 + (long)tapBase * 256, ldsBase);

    v8f_t acc = {};
    int tapIdx = 0;
    for (int ky = ks * 3; ky < ks * 3 + 3; ++ky) {
        for (int kx = 0; kx < 9; ++kx) {
            int cur = tapIdx & 1;
            if (wid == 0) {
                if (tapIdx + 1 < 27) {
                    tdm_load_tile(Btap0 + (long)(tapBase + tapIdx + 1) * 256,
                                  ldsBase + (unsigned)(((tapIdx + 1) & 1) * 8192));
                    __builtin_amdgcn_s_wait_tensorcnt(1);   // current tile done
                } else {
                    __builtin_amdgcn_s_wait_tensorcnt(0);
                }
            }
            __syncthreads();                                // B tile visible to block

            const _Float16* aTap = aBase + (ky * 56 + kx) * 256;
            const _Float16* bsm  = &smemB[cur][0] + ln * 256 + g * 8;
            __builtin_prefetch(aTap + 256, 0, 1);           // next-tap A stream
#pragma unroll
            for (int kk = 0; kk < 8; ++kk) {
                const v8h_t* pa = (const v8h_t*)(aTap + kk * 32);
                v8h_t a0 = pa[0];                           // K rel g*8..+7
                v8h_t a1 = pa[2];                           // K rel 16+g*8..+7
                v16h_t a;
#pragma unroll
                for (int i = 0; i < 8; i++) { a[i] = a0[i]; a[8 + i] = a1[i]; }
                const v8h_t* pb = (const v8h_t*)(bsm + kk * 32);
                v8h_t b0 = pb[0];                           // K rel g*8..+7
                v8h_t b1 = pb[2];                           // K rel 16+g*8..+7
                v16h_t bf;
#pragma unroll
                for (int i = 0; i < 8; i++) { bf[i] = b0[i]; bf[8 + i] = b1[i]; }
                acc = __builtin_amdgcn_wmma_f32_16x16x32_f16(
                    false, a, false, bf, (short)0, acc, false, false);
            }
            __syncthreads();                                // safe to overwrite buf
            tapIdx++;
        }
    }
    // C layout: VGPR rr -> M = mt*16 + g*8 + rr, N = oc
    int oc = nt * 16 + ln;
    int mrow0 = mt * 16 + g * 8;
#pragma unroll
    for (int rr = 0; rr < 8; rr++)
        O2p[((long)ks * M2 + mrow0 + rr) * 256 + oc] = acc[rr];
}

// ---------------- K-partial reduce + bias + squash of primary capsules ----------
__global__ void squash_prim(const float* __restrict__ O2p, const float* __restrict__ b2,
                            float* __restrict__ prim) {
    int idx = blockIdx.x * 256 + threadIdx.x;         // 2*18432
    if (idx >= 2 * P_ROUTES) return;
    int b = idx / P_ROUTES, p = idx % P_ROUTES;
    int type = p / HW, s = p % HW;
    long mrow = (long)(b * HW + s) * 256 + type * 8;
    float vv[8], sq = 0.f;
#pragma unroll
    for (int d = 0; d < 8; d++) {
        float u = O2p[mrow + d] + O2p[(long)M2 * 256 + mrow + d]
                + O2p[(long)2 * M2 * 256 + mrow + d] + b2[type * 8 + d];
        vv[d] = u; sq += u * u;
    }
    float scale = sq / (1.f + sq) / sqrtf(sq + 1e-8f);
#pragma unroll
    for (int d = 0; d < 8; d++) prim[(long)idx * 8 + d] = scale * vv[d];
}

// ---------------- routing (u_hat eliminated algebraically) ----------------

// wv[b][t][cc][d] = sum_e Wf[t][d][cc*16+e] * v[b][cc*16+e]
__global__ void k_wv(const float* __restrict__ Wf, const float* __restrict__ vbuf,
                     float* __restrict__ wv) {
    int idx = blockIdx.x * 256 + threadIdx.x;         // 2*32*276*8
    int d = idx & 7, r = idx >> 3;
    int cc = r % NUM_CLASSES; int r2 = r / NUM_CLASSES;
    int t = r2 % 32, b = r2 / 32;
    const float* wr = Wf + ((long)t * 8 + d) * CE + cc * 16;
    const float* vr = vbuf + (long)b * CE + cc * 16;
    float a = 0.f;
#pragma unroll
    for (int e = 0; e < 16; e++) a = fmaf(wr[e], vr[e], a);
    wv[idx] = a;
}

// per route: b_log += prim . wv (it>0); then softmax stats (max, Z) -> rowstat
__global__ __launch_bounds__(64) void k_blog(const float* __restrict__ prim,
                                             const float* __restrict__ wv,
                                             float* __restrict__ b_log,
                                             float* __restrict__ rowstat, int it) {
    int bid = blockIdx.x;                  // 2*32*9
    int b = bid / 288, r1 = bid % 288;
    int t = r1 / 9, chunk = r1 % 9;
    int q = chunk * 64 + threadIdx.x;      // 0..575 within (b,t)
    int type = q / 18, j = q % 18;
    int p = type * HW + t + 32 * j;
    long bp = (long)b * P_ROUTES + p;
    float pr[8];
#pragma unroll
    for (int d = 0; d < 8; d++) pr[d] = prim[bp * 8 + d];
    const float* wv0 = wv + ((long)(b * 32 + t)) * NUM_CLASSES * 8;
    float* bl = b_log + bp * NUM_CLASSES;
    float m = -1e30f;
    for (int cc = 0; cc < NUM_CLASSES; cc++) {
        float v = bl[cc];
        if (it > 0) {
            const float* wr = wv0 + cc * 8;   // uniform per block: scalar broadcast
            float du = 0.f;
#pragma unroll
            for (int d = 0; d < 8; d++) du = fmaf(pr[d], wr[d], du);
            v += du; bl[cc] = v;
        }
        m = fmaxf(m, v);
    }
    float Z = 0.f;
    for (int cc = 0; cc < NUM_CLASSES; cc++) Z += __expf(bl[cc] - m);
    rowstat[bp * 2] = m; rowstat[bp * 2 + 1] = Z;
}

// g[b][t][cc][d] = sum_{p in (b,t)} c[p,cc] * prim[p,d]
__global__ __launch_bounds__(64) void k_g(const float* __restrict__ prim,
                                          const float* __restrict__ b_log,
                                          const float* __restrict__ rowstat,
                                          float* __restrict__ gbuf) {
    __shared__ float red[64 * 8];
    int blk = blockIdx.x;                  // 2*32*276
    int b = blk / (32 * NUM_CLASSES);
    int rem = blk % (32 * NUM_CLASSES);
    int t = rem / NUM_CLASSES, cc = rem % NUM_CLASSES;
    float acc[8] = {0, 0, 0, 0, 0, 0, 0, 0};
    for (int q = threadIdx.x; q < HW; q += 64) {
        int type = q / 18, j = q % 18;
        int p = type * HW + t + 32 * j;
        long bp = (long)b * P_ROUTES + p;
        float mm = rowstat[bp * 2], Z = rowstat[bp * 2 + 1];
        float c = __expf(b_log[bp * NUM_CLASSES + cc] - mm) / Z;
#pragma unroll
        for (int d = 0; d < 8; d++) acc[d] = fmaf(c, prim[bp * 8 + d], acc[d]);
    }
#pragma unroll
    for (int d = 0; d < 8; d++) red[threadIdx.x * 8 + d] = acc[d];
    __syncthreads();
    for (int off = 32; off >= 1; off >>= 1) {
        if (threadIdx.x < off)
#pragma unroll
            for (int d = 0; d < 8; d++)
                red[threadIdx.x * 8 + d] += red[(threadIdx.x + off) * 8 + d];
        __syncthreads();
    }
    if (threadIdx.x < 8)
        gbuf[(((long)(b * 32 + t)) * NUM_CLASSES + cc) * 8 + threadIdx.x] = red[threadIdx.x];
}

// s[b][ce] = sum_{t,d} g[b][t][cc][d] * Wf[t][d][ce]
__global__ void k_s(const float* __restrict__ gbuf, const float* __restrict__ Wf,
                    float* __restrict__ sbuf) {
    int idx = blockIdx.x * 256 + threadIdx.x;         // 2*4416
    if (idx >= 2 * CE) return;
    int b = idx / CE, ce = idx % CE;
    int cc = ce >> 4;
    float a = 0.f;
    for (int t = 0; t < 32; t++)
#pragma unroll
        for (int d = 0; d < 8; d++)
            a = fmaf(gbuf[(((long)(b * 32 + t)) * NUM_CLASSES + cc) * 8 + d],
                     Wf[((long)t * 8 + d) * CE + ce], a);
    sbuf[idx] = a;
}

// v = squash(s); on last iter also write class lengths to out
__global__ void k_v(const float* __restrict__ sbuf, float* __restrict__ vbuf,
                    float* __restrict__ out, int write_out) {
    int idx = blockIdx.x * 256 + threadIdx.x;         // 2*276
    if (idx >= 2 * NUM_CLASSES) return;
    const float* sr = sbuf + (long)idx * 16;
    float tmp[16], sq = 0.f;
#pragma unroll
    for (int e = 0; e < 16; e++) { tmp[e] = sr[e]; sq += tmp[e] * tmp[e]; }
    float scale = sq / (1.f + sq) / sqrtf(sq + 1e-8f);
    float* vr = vbuf + (long)idx * 16;
#pragma unroll
    for (int e = 0; e < 16; e++) vr[e] = scale * tmp[e];
    if (write_out) out[idx] = scale * sqrtf(sq);
}

// ---------------- launch ----------------
static inline char* bump(char*& p, size_t bytes) {
    char* r = p;
    p += (bytes + 255) & ~(size_t)255;
    return r;
}

extern "C" void kernel_launch(void* const* d_in, const int* in_sizes, int n_in,
                              void* d_out, int out_size, void* d_ws, size_t ws_size,
                              hipStream_t stream) {
    const float* x   = (const float*)d_in[0];
    const float* w1  = (const float*)d_in[1];
    const float* b1  = (const float*)d_in[2];
    const float* w2  = (const float*)d_in[3];
    const float* b2  = (const float*)d_in[4];
    const float* W   = (const float*)d_in[5];
    float* out = (float*)d_out;

    char* p = (char*)d_ws;
    float*    w1t     = (float*)   bump(p, (size_t)81 * 256 * 4);
    _Float16* H1      = (_Float16*)bump(p, (size_t)2 * 56 * 56 * 256 * 2);
    _Float16* Bpack   = (_Float16*)bump(p, (size_t)256 * K2 * 2);
    float*    O2p     = (float*)   bump(p, (size_t)3 * M2 * 256 * 4);   // K-split partials
    float*    prim    = (float*)   bump(p, (size_t)2 * P_ROUTES * 8 * 4);
    float*    Wf      = (float*)   bump(p, (size_t)32 * 8 * CE * 4);
    float*    b_log   = (float*)   bump(p, (size_t)2 * P_ROUTES * NUM_CLASSES * 4);
    float*    wv      = (float*)   bump(p, (size_t)2 * 32 * NUM_CLASSES * 8 * 4);
    float*    rowstat = (float*)   bump(p, (size_t)2 * P_ROUTES * 2 * 4);
    float*    gbuf    = (float*)   bump(p, (size_t)2 * 32 * NUM_CLASSES * 8 * 4);
    float*    sbuf    = (float*)   bump(p, (size_t)2 * CE * 4);
    float*    vbuf    = (float*)   bump(p, (size_t)2 * CE * 4);

    pack_w1t<<<81, 256, 0, stream>>>(w1, w1t);
    pack_w2<<<K2, 256, 0, stream>>>(w2, Bpack);
    pack_Wf<<<4416, 256, 0, stream>>>(W, Wf);
    conv1_relu<<<2 * 3136, 256, 0, stream>>>(x, w1t, b1, H1);
    conv2_wmma<<<864, 128, 0, stream>>>(H1, Bpack, O2p);  // 16 nt * 18 mtg * 3 ks
    squash_prim<<<144, 256, 0, stream>>>(O2p, b2, prim);

    (void)hipMemsetAsync(b_log, 0, (size_t)2 * P_ROUTES * NUM_CLASSES * 4, stream);

    for (int it = 0; it < 3; it++) {
        if (it > 0)
            k_wv<<<552, 256, 0, stream>>>(Wf, vbuf, wv);
        k_blog<<<576, 64, 0, stream>>>(prim, wv, b_log, rowstat, it);
        k_g<<<2 * 32 * NUM_CLASSES, 64, 0, stream>>>(prim, b_log, rowstat, gbuf);
        k_s<<<35, 256, 0, stream>>>(gbuf, Wf, sbuf);
        k_v<<<3, 256, 0, stream>>>(sbuf, vbuf, out, it == 2 ? 1 : 0);
    }
}
